// SpatialAttentionGate_76845554860820
// MI455X (gfx1250) — matrix operand order repaired and checked by reading, not compiled, not run on hardware
//
#include <hip/hip_runtime.h>
#include <hip/hip_bf16.h>

typedef _Float16 f16_t;
typedef __attribute__((ext_vector_type(16))) _Float16 v16h;
typedef __attribute__((ext_vector_type(8)))  _Float16 v8h;
typedef __attribute__((ext_vector_type(8)))  float    v8f;

#define NPIX 4096
#define CCH  512
#define CQD  64
#define NB   8

static __device__ __forceinline__ v8f wmma_f16(v16h a, v16h b, v8f c) {
  // D = A(16x32 f16) * B(32x16 f16) + C(16x16 f32)
  return __builtin_amdgcn_wmma_f32_16x16x32_f16(
      /*neg_a=*/false, a, /*neg_b=*/false, b,
      /*c_mod=*/(short)0, c, /*reuse_a=*/false, /*reuse_b=*/false);
}

// Lane-xor fetch within a 16-lane row via DPP ROW_XMASK (VALU, not DS unit).
template <int MASK>
static __device__ __forceinline__ float dpp_xor(float v) {
  int i = __builtin_bit_cast(int, v);
  int p = __builtin_amdgcn_update_dpp(i, i, 0x160 | MASK, 0xF, 0xF, true);
  return __builtin_bit_cast(float, p);
}

// Butterfly reductions over the 16 lanes of each half-wave; results are
// replicated across all 16 lanes of the half.
static __device__ __forceinline__ void rowmax_reduce8(float (&v)[8]) {
  #pragma unroll
  for (int r = 0; r < 8; ++r) v[r] = fmaxf(v[r], dpp_xor<1>(v[r]));
  #pragma unroll
  for (int r = 0; r < 8; ++r) v[r] = fmaxf(v[r], dpp_xor<2>(v[r]));
  #pragma unroll
  for (int r = 0; r < 8; ++r) v[r] = fmaxf(v[r], dpp_xor<4>(v[r]));
  #pragma unroll
  for (int r = 0; r < 8; ++r) v[r] = fmaxf(v[r], dpp_xor<8>(v[r]));
}
static __device__ __forceinline__ void rowsum_reduce8(float (&v)[8]) {
  #pragma unroll
  for (int r = 0; r < 8; ++r) v[r] += dpp_xor<1>(v[r]);
  #pragma unroll
  for (int r = 0; r < 8; ++r) v[r] += dpp_xor<2>(v[r]);
  #pragma unroll
  for (int r = 0; r < 8; ++r) v[r] += dpp_xor<4>(v[r]);
  #pragma unroll
  for (int r = 0; r < 8; ++r) v[r] += dpp_xor<8>(v[r]);
}

// ---------------------------------------------------------------------------
// Stage 1: Y = W @ x_b + bias, output f16, optionally transposed.
// ---------------------------------------------------------------------------
template <int M, bool TRANS>
__global__ __launch_bounds__(64) void proj_wmma_kernel(
    const float* __restrict__ W, const float* __restrict__ bias,
    const float* __restrict__ x, f16_t* __restrict__ Y)
{
  const int lane = threadIdx.x & 31;
  const int wave = threadIdx.x >> 5;
  const int n0 = (blockIdx.x * 2 + wave) * 16;
  const int m0 = blockIdx.y * 16;
  const int b  = blockIdx.z;

  const float* xb = x + (size_t)b * CCH * NPIX;
  f16_t*       Yb = Y + (size_t)b * M * NPIX;

  const int col = lane & 15;
  const int hi  = lane >> 4;

  const float* wrow = W  + (size_t)(m0 + col) * CCH;
  const float* xcol = xb + n0 + col;

  v8f acc = {};
  for (int c0 = 0; c0 < CCH; c0 += 32) {
    v16h a, bf;
    #pragma unroll
    for (int t = 0; t < 16; ++t) {
      int ka = (hi << 4) + t;                           // A layout
      a[t] = (f16_t)wrow[c0 + ka];
    }
    #pragma unroll
    for (int t = 0; t < 16; ++t) {
      int kb = (hi << 4) + t;                           // B layout
      bf[t] = (f16_t)xcol[(size_t)(c0 + kb) * NPIX];
    }
    acc = wmma_f16(a, bf, acc);
  }

  if constexpr (TRANS) {
    v8h hv;
    #pragma unroll
    for (int r = 0; r < 8; ++r)
      hv[r] = (f16_t)(acc[r] + bias[m0 + (hi << 3) + r]);
    *(v8h*)(Yb + (size_t)(n0 + col) * M + m0 + (hi << 3)) = hv;
  } else {
    #pragma unroll
    for (int r = 0; r < 8; ++r) {
      int m = m0 + r + (hi << 3);
      Yb[(size_t)m * NPIX + n0 + col] = (f16_t)(acc[r] + bias[m]);
    }
  }
}

// ---------------------------------------------------------------------------
// Stage 2: cooperative fused flash-attention + residual.
// Block = 16 waves = one (batch, 16-query tile). Chunk = 256 keys.
//   - wave w computes S tile for keys [j0+16w, j0+16w+16)  (2 WMMA, no dup)
//   - row stats merged across waves via LDS [row][wave] + DPP butterflies
//   - shared P (16x256 f16, LDS) consumed by all waves
//   - wave w owns channels [32w, 32w+32): 16 O-WMMAs per chunk
// ---------------------------------------------------------------------------
__global__ __launch_bounds__(512) void attn_wmma_kernel(
    const f16_t* __restrict__ Qt, const f16_t* __restrict__ Kt,
    const f16_t* __restrict__ V,  const float* __restrict__ x,
    const float* __restrict__ gamma, float* __restrict__ out)
{
  __shared__ f16_t p_lds[16 * 256];     // P tile [i][j], 8KB
  __shared__ float stA[16 * 16];        // per-wave row maxes [row][wave]
  __shared__ float stB[16 * 16];        // per-wave row sums  [row][wave]
  __shared__ float sl[16];              // per-row alpha / lsum broadcast

  const int lane = threadIdx.x & 31;
  const int wave = threadIdx.x >> 5;    // 0..15
  const int i0 = blockIdx.x * 16;
  const int b  = blockIdx.y;
  const int c0 = wave * 32;

  const f16_t* qt = Qt + (size_t)b * NPIX * CQD;
  const f16_t* kt = Kt + (size_t)b * NPIX * CQD;
  const f16_t* vb = V  + (size_t)b * CCH * NPIX;

  const int col = lane & 15;
  const int hi  = lane >> 4;

  // Q A-fragments (d = 0..63, two K=32 chunks) — resident all loop long.
  v16h aq0, aq1;
  const f16_t* qrow = qt + (size_t)(i0 + col) * CQD;
  #pragma unroll
  for (int t = 0; t < 16; ++t) {
    int d = (hi << 4) + t;                           // A layout
    aq0[t] = qrow[d];
    aq1[t] = qrow[32 + d];
  }

  v8f o0 = {}, o1 = {};
  float rmax[8], rsum[8];
  #pragma unroll
  for (int r = 0; r < 8; ++r) { rmax[r] = -1e30f; rsum[r] = 0.0f; }

  const f16_t* vrow0 = vb + (size_t)(c0 + col)      * NPIX;
  const f16_t* vrow1 = vb + (size_t)(c0 + 16 + col) * NPIX;

  for (int j0 = 0; j0 < NPIX; j0 += 256) {
    const int jw = j0 + wave * 16;      // this wave's 16 key columns

    // ---- this wave's K B-fragments (contiguous 32B/lane) ----
    v16h bk0, bk1;
    const f16_t* krow = kt + (size_t)(jw + col) * CQD;
    #pragma unroll
    for (int t = 0; t < 16; ++t) {
      int d = (hi << 4) + t;                         // B layout
      bk0[t] = krow[d];
      bk1[t] = krow[32 + d];
    }
    // prefetch next chunk (gfx1250 global_prefetch_b8)
    if (j0 + 256 < NPIX) {
      __builtin_prefetch(kt + (size_t)(jw + 256 + col) * CQD, 0, 3);
      __builtin_prefetch(vrow0 + j0 + 256 + hi * 256, 0, 3);
      __builtin_prefetch(vrow1 + j0 + 256 + hi * 256, 0, 3);
    }

    // ---- S tile: 2 WMMAs (unique work per wave) ----
    v8f s = {};
    s = wmma_f16(aq0, bk0, s);
    s = wmma_f16(aq1, bk1, s);

    // ---- phase 1: per-tile row max -> LDS [row][wave] ----
    float tm[8];
    #pragma unroll
    for (int r = 0; r < 8; ++r) tm[r] = s[r];
    rowmax_reduce8(tm);
    if (lane == 0 || lane == 16) {
      #pragma unroll
      for (int r = 0; r < 8; ++r) stA[(r + (hi << 3)) * 16 + wave] = tm[r];
    }
    __syncthreads();

    // ---- phase 2: global row max, exp, P store, per-tile row sum ----
    float nm[8];
    #pragma unroll
    for (int r = 0; r < 8; ++r)
      nm[r] = fmaxf(rmax[r], stA[(r + (hi << 3)) * 16 + col]);
    rowmax_reduce8(nm);                 // reduce over waves (cols)

    float alpha[8], ps[8];
    #pragma unroll
    for (int r = 0; r < 8; ++r) {
      alpha[r] = __expf(rmax[r] - nm[r]);
      float p = __expf(s[r] - nm[r]);
      p_lds[(r + (hi << 3)) * 256 + wave * 16 + col] = (f16_t)p;
      ps[r] = p;
    }
    rowsum_reduce8(ps);                 // per-tile sums, replicated
    if (lane == 0 || lane == 16) {
      #pragma unroll
      for (int r = 0; r < 8; ++r) stB[(r + (hi << 3)) * 16 + wave] = ps[r];
    }
    if (wave == 0 && (lane == 0 || lane == 16)) {
      #pragma unroll
      for (int r = 0; r < 8; ++r) sl[r + (hi << 3)] = alpha[r];
    }
    __syncthreads();

    // ---- phase 3: global row sum, rescale, O accumulate ----
    float gs[8];
    #pragma unroll
    for (int r = 0; r < 8; ++r) gs[r] = stB[(r + (hi << 3)) * 16 + col];
    rowsum_reduce8(gs);                 // reduce over waves (cols)
    #pragma unroll
    for (int r = 0; r < 8; ++r) {
      rsum[r] = rsum[r] * alpha[r] + gs[r];
      rmax[r] = nm[r];
    }
    float acol = sl[col];
    #pragma unroll
    for (int r = 0; r < 8; ++r) { o0[r] *= acol; o1[r] *= acol; }

    #pragma unroll
    for (int q = 0; q < 8; ++q) {       // 8 K=32 chunks over 256 keys
      v16h bp;                          // P^T B-frag: B[k=j][n=i]=P[i][j]
      #pragma unroll
      for (int t = 0; t < 16; ++t)
        bp[t] = p_lds[col * 256 + 32 * q + (hi << 4) + t];  // 32B/lane
      v16h av0q, av1q;                  // V A-frags (16 contiguous)
      #pragma unroll
      for (int t = 0; t < 16; ++t) {
        int k = (hi << 4) + t;                              // A layout
        av0q[t] = vrow0[j0 + 32 * q + k];
        av1q[t] = vrow1[j0 + 32 * q + k];
      }
      o0 = wmma_f16(av0q, bp, o0);
      o1 = wmma_f16(av1q, bp, o1);
    }
    __syncthreads();                    // protect p_lds/stA/sl for next chunk
  }

  // final normalization: publish row sums (identical in all waves)
  if (wave == 0 && (lane == 0 || lane == 16)) {
    #pragma unroll
    for (int r = 0; r < 8; ++r) sl[r + (hi << 3)] = rsum[r];
  }
  __syncthreads();
  float linv = 1.0f / sl[col];
  float g = gamma[0];

  const float* xb = x + (size_t)b * CCH * NPIX;
  float* ob = out + (size_t)b * CCH * NPIX;
  #pragma unroll
  for (int r = 0; r < 8; ++r) {
    int m = r + (hi << 3);
    int i = i0 + col;
    size_t idx0 = (size_t)(c0 + m)      * NPIX + i;
    size_t idx1 = (size_t)(c0 + 16 + m) * NPIX + i;
    ob[idx0] = g * (o0[r] * linv) + xb[idx0];
    ob[idx1] = g * (o1[r] * linv) + xb[idx1];
  }
}

// ---------------------------------------------------------------------------
extern "C" void kernel_launch(void* const* d_in, const int* in_sizes, int n_in,
                              void* d_out, int out_size, void* d_ws, size_t ws_size,
                              hipStream_t stream) {
  const float* x     = (const float*)d_in[0];
  const float* Wq    = (const float*)d_in[1];
  const float* bq    = (const float*)d_in[2];
  const float* Wk    = (const float*)d_in[3];
  const float* bk    = (const float*)d_in[4];
  const float* Wv    = (const float*)d_in[5];
  const float* bv    = (const float*)d_in[6];
  const float* gamma = (const float*)d_in[7];
  float* out = (float*)d_out;

  // workspace: Qt [B,N,CQ] f16 | Kt [B,N,CQ] f16 | V [B,C,N] f16  (40 MB)
  f16_t* Qt = (f16_t*)d_ws;
  f16_t* Kt = Qt + (size_t)NB * NPIX * CQD;
  f16_t* Vb = Kt + (size_t)NB * NPIX * CQD;

  dim3 blk1(64, 1, 1);
  dim3 grdQK(NPIX / 32, CQD / 16, NB);
  hipLaunchKernelGGL((proj_wmma_kernel<CQD, true>), grdQK, blk1, 0, stream,
                     Wq, bq, x, Qt);
  hipLaunchKernelGGL((proj_wmma_kernel<CQD, true>), grdQK, blk1, 0, stream,
                     Wk, bk, x, Kt);
  dim3 grdV(NPIX / 32, CCH / 16, NB);
  hipLaunchKernelGGL((proj_wmma_kernel<CCH, false>), grdV, blk1, 0, stream,
                     Wv, bv, x, Vb);

  dim3 blk2(512, 1, 1);
  dim3 grd2(NPIX / 16, NB, 1);          // (256, 8): one block per (itile, b)
  hipLaunchKernelGGL(attn_wmma_kernel, grd2, blk2, 0, stream,
                     Qt, Kt, Vb, x, gamma, out);
}
